// KnowledgeAwareVariationalAutoEncoder_27771258536335
// MI455X (gfx1250) — matrix operand order repaired and checked by reading, not compiled, not run on hardware
//
#include <hip/hip_runtime.h>
#include <hip/hip_bf16.h>

#define BB 512
#define NN 50000
#define DD 64
#define HH 600
#define LL 200
#define ALPHAF 1.0f

#define SPLITK_P 25
#define SPLITK_E 8
#define POOL_STRIDE 80
#define HP_STRIDE 608

#define CHUNK_E 6272  /* ceil(50000/8) rounded up to mult of 32 */
#define CHUNK_P 2016  /* ceil(50000/25) rounded up to mult of 32 */

#define BLDN 132      /* enc/dec B LDS row stride (f32), mult of 4 */
#define PLDN 68       /* pool    B LDS row stride (f32), mult of 4 */

typedef __attribute__((ext_vector_type(16))) __bf16 bf16x16;
typedef __attribute__((ext_vector_type(8)))  float  f32x8;
typedef __attribute__((ext_vector_type(4)))  int    v4i;

// ---------------- CDNA5 async global->LDS path (guarded) ---------------------

#if defined(__has_builtin)
#if __has_builtin(__builtin_amdgcn_global_load_async_to_lds_b128) && \
    __has_builtin(__builtin_amdgcn_s_wait_asynccnt)
#define HAVE_ASYNC 1
#endif
#endif
#ifndef HAVE_ASYNC
#define HAVE_ASYNC 0
#endif

#if HAVE_ASYNC
__device__ __forceinline__ void async_ld_b128(const void* g, void* l) {
  __builtin_amdgcn_global_load_async_to_lds_b128(
      (__attribute__((address_space(1))) v4i*)g,
      (__attribute__((address_space(3))) v4i*)l, 0, 0);
}
#endif

// ---------------- WMMA fragment gathers (contiguous k order, same for A and B) -------
// LDS tiles held in f32; convert to bf16 at gather time (VALU co-executes
// with XDL WMMA ops per ISA 8.4).

// A 16x32: lanes 0-15: M=lane, V0-7 -> K0..15; lanes 16-31: K16..31
__device__ __forceinline__ bf16x16 frag_a_f32(const float* As, int lane) {
  int m = lane & 15;
  int kh = (lane >> 4) << 4;
  bf16x16 a;
#pragma unroll
  for (int i = 0; i < 8; ++i) {
    int kb = 2 * i + kh;
    a[2 * i]     = (__bf16)As[m * 32 + kb];
    a[2 * i + 1] = (__bf16)As[m * 32 + kb + 1];
  }
  return a;
}

// B 32x16: lanes 0-15: N=lane, V0-7 -> K0..15; lanes 16-31: K16..31.
// Bs is row-major [k][n] (f32, row stride ldn).
__device__ __forceinline__ bf16x16 frag_b_f32(const float* Bs, int ldn, int lane,
                                              int n0) {
  int n = n0 + (lane & 15);
  int kh = (lane >> 4) << 4;
  bf16x16 b;
#pragma unroll
  for (int i = 0; i < 8; ++i) {
    int kb = 2 * i + kh;
    b[2 * i]     = (__bf16)Bs[kb * ldn + n];
    b[2 * i + 1] = (__bf16)Bs[(kb + 1) * ldn + n];
  }
  return b;
}

// ---------------- 1) pooled = inputs @ item_emb  (+ row counts) --------------

__global__ __launch_bounds__(128) void k_pool(const float* __restrict__ inputs,
                                              const float* __restrict__ item_emb,
                                              float* __restrict__ poolPart) {
  __shared__ alignas(16) float Asb[2][16 * 32];
  __shared__ alignas(16) float Bsb[2][32 * PLDN];
  const int rowT = blockIdx.x;      // 0..31
  const int s    = blockIdx.y;      // 0..SPLITK_P-1
  const int row0 = rowT * 16;
  const int tid = threadIdx.x, lane = tid & 31, wave = tid >> 5;
  const int ks = s * CHUNK_P;
  const int ke = min(ks + CHUNK_P, NN);
  const int len = ke - ks;
  const int nfull = len >> 5;
  const int tail = len & 31;

  f32x8 acc = {};
  f32x8 accCnt = {};
  bf16x16 ones;
#pragma unroll
  for (int i = 0; i < 16; ++i) ones[i] = (__bf16)1.0f;

  auto do_mma = [&](const float* Af, const float* Bf) {
    bf16x16 a = frag_a_f32(Af, lane);
    bf16x16 b = frag_b_f32(Bf, PLDN, lane, wave * 16);
    acc = __builtin_amdgcn_wmma_f32_16x16x32_bf16(false, a, false, b, (short)0,
                                                  acc, false, false);
    if (wave == 0)
      accCnt = __builtin_amdgcn_wmma_f32_16x16x32_bf16(false, a, false, ones,
                                                       (short)0, accCnt, false, false);
  };

#if HAVE_ASYNC
  auto issue = [&](int buf, int k0) {
    int r = tid >> 3, c4 = (tid & 7) * 4;
    async_ld_b128(inputs + (size_t)(row0 + r) * NN + k0 + c4, &Asb[buf][r * 32 + c4]);
#pragma unroll
    for (int q = 0; q < 4; ++q) {
      int idx = q * 128 + tid;
      int kk = idx >> 4, cc = (idx & 15) * 4;
      async_ld_b128(item_emb + (size_t)(k0 + kk) * DD + cc, &Bsb[buf][kk * PLDN + cc]);
    }
  };
  if (nfull > 0) issue(0, ks);
  for (int i = 0; i < nfull; ++i) {
    __syncthreads();  // protect buffer (i+1)&1 from overwrite-before-consumed
    if (i + 1 < nfull) {
      issue((i + 1) & 1, ks + (i + 1) * 32);
      __builtin_amdgcn_s_wait_asynccnt(5);  // 5 async ops per tile per wave
    } else {
      __builtin_amdgcn_s_wait_asynccnt(0);
    }
    __syncthreads();
    do_mma(Asb[i & 1], Bsb[i & 1]);
  }
  if (tail) {
    const int k0 = ks + nfull * 32;
    __syncthreads();
    for (int idx = tid; idx < 16 * 32; idx += 128) {
      int r = idx >> 5, c = idx & 31, gk = k0 + c;
      Asb[0][idx] = (gk < ke) ? inputs[(size_t)(row0 + r) * NN + gk] : 0.0f;
    }
    for (int idx = tid; idx < 32 * 64; idx += 128) {
      int kk = idx >> 6, c = idx & 63, gk = k0 + kk;
      Bsb[0][kk * PLDN + c] = (gk < ke) ? item_emb[(size_t)gk * DD + c] : 0.0f;
    }
    __syncthreads();
    do_mma(Asb[0], Bsb[0]);
  }
#else
  for (int k0 = ks; k0 < ke; k0 += 32) {
    __syncthreads();
    for (int idx = tid; idx < 16 * 32; idx += 128) {
      int r = idx >> 5, c = idx & 31, gk = k0 + c;
      Asb[0][idx] = (gk < ke) ? inputs[(size_t)(row0 + r) * NN + gk] : 0.0f;
    }
    for (int idx = tid; idx < 32 * 64; idx += 128) {
      int kk = idx >> 6, c = idx & 63, gk = k0 + kk;
      Bsb[0][kk * PLDN + c] = (gk < ke) ? item_emb[(size_t)gk * DD + c] : 0.0f;
    }
    __syncthreads();
    do_mma(Asb[0], Bsb[0]);
  }
#endif

  float* outp = poolPart + (size_t)s * BB * POOL_STRIDE;
  const int col = wave * 16 + (lane & 15);
  const int rbase = (lane >> 4) * 8;
#pragma unroll
  for (int v = 0; v < 8; ++v)
    outp[(size_t)(row0 + rbase + v) * POOL_STRIDE + col] = acc[v];
  if (wave == 0 && (lane & 15) == 0) {
#pragma unroll
    for (int v = 0; v < 8; ++v)
      outp[(size_t)(row0 + rbase + v) * POOL_STRIDE + 64] = accCnt[v];
  }
}

// ---------------- 2) finalize pooled: p = a*l2n(pooled/cnt), xinv ------------

__global__ __launch_bounds__(64) void k_pool_fin(const float* __restrict__ poolPart,
                                                 float* __restrict__ p,
                                                 float* __restrict__ xinv) {
  const int b = blockIdx.x;
  const int d = threadIdx.x;  // 64 threads
  float s = 0.0f, cnt = 0.0f;
  for (int i = 0; i < SPLITK_P; ++i) {
    const float* pp = poolPart + ((size_t)i * BB + b) * POOL_STRIDE;
    s += pp[d];
    cnt += pp[64];
  }
  float cntc = fmaxf(cnt, 1.0f);
  float pooled = s / cntc;
  __shared__ float red[64];
  red[d] = pooled * pooled;
  __syncthreads();
  for (int off = 32; off > 0; off >>= 1) {
    if (d < off) red[d] += red[d + off];
    __syncthreads();
  }
  float ssq = red[0];
  float inv = rsqrtf(fmaxf(ssq, 1e-12f));
  p[(size_t)b * DD + d] = ALPHAF * (pooled * inv);
  if (d == 0) {
    float pn2 = (ALPHAF * ALPHAF) * (ssq / fmaxf(ssq, 1e-12f));
    xinv[b] = rsqrtf(fmaxf(cnt + pn2, 1e-12f));  // 1/||x||
  }
}

// ---------------- 3) encoder GEMM: hpart[s] = inputs @ W1[0:N,:] -------------

__global__ __launch_bounds__(128) void k_enc_gemm(const float* __restrict__ inputs,
                                                  const float* __restrict__ W1,
                                                  float* __restrict__ hpart) {
  __shared__ alignas(16) float Asb[2][16 * 32];
  __shared__ alignas(16) float Bsb[2][32 * BLDN];
  const int rowT = blockIdx.x;  // fastest -> concurrent blocks share W1 tile in L2
  const int cb   = blockIdx.y;
  const int s    = blockIdx.z;
  const int row0 = rowT * 16, c0 = cb * 128;
  const int tid = threadIdx.x, lane = tid & 31, wave = tid >> 5;
  const int ks = s * CHUNK_E;
  const int ke = min(ks + CHUNK_E, NN);
  const int len = ke - ks;
  const int nfull = len >> 5;
  const int tail = len & 31;

  f32x8 acc0 = {}, acc1 = {};
  auto do_mma = [&](const float* Af, const float* Bf) {
    bf16x16 a  = frag_a_f32(Af, lane);
    bf16x16 b0 = frag_b_f32(Bf, BLDN, lane, wave * 32);
    bf16x16 b1 = frag_b_f32(Bf, BLDN, lane, wave * 32 + 16);
    acc0 = __builtin_amdgcn_wmma_f32_16x16x32_bf16(false, a, false, b0, (short)0,
                                                   acc0, false, false);
    acc1 = __builtin_amdgcn_wmma_f32_16x16x32_bf16(false, a, false, b1, (short)0,
                                                   acc1, false, false);
  };

#if HAVE_ASYNC
  // NOTE: column-edge over-reads (c0+127 > 599) stay inside the W1 allocation
  // (rows here are < 50000 of 50064); garbage only feeds unstored columns.
  auto issue = [&](int buf, int k0) {
    int r = tid >> 3, c4 = (tid & 7) * 4;
    async_ld_b128(inputs + (size_t)(row0 + r) * NN + k0 + c4, &Asb[buf][r * 32 + c4]);
#pragma unroll
    for (int q = 0; q < 8; ++q) {
      int idx = q * 128 + tid;
      int kk = idx >> 5, cc = (idx & 31) * 4;
      async_ld_b128(W1 + (size_t)(k0 + kk) * HH + c0 + cc, &Bsb[buf][kk * BLDN + cc]);
    }
  };
  if (nfull > 0) issue(0, ks);
  for (int i = 0; i < nfull; ++i) {
    __syncthreads();
    if (i + 1 < nfull) {
      issue((i + 1) & 1, ks + (i + 1) * 32);
      __builtin_amdgcn_s_wait_asynccnt(9);  // 9 async ops per tile per wave
    } else {
      __builtin_amdgcn_s_wait_asynccnt(0);
    }
    __syncthreads();
    do_mma(Asb[i & 1], Bsb[i & 1]);
  }
  if (tail) {
    const int k0 = ks + nfull * 32;
    __syncthreads();
    for (int idx = tid; idx < 16 * 32; idx += 128) {
      int r = idx >> 5, c = idx & 31, gk = k0 + c;
      Asb[0][idx] = (gk < ke) ? inputs[(size_t)(row0 + r) * NN + gk] : 0.0f;
    }
    for (int idx = tid; idx < 32 * 128; idx += 128) {
      int kk = idx >> 7, c = idx & 127, gk = k0 + kk, gc = c0 + c;
      Bsb[0][kk * BLDN + c] = (gk < ke && gc < HH) ? W1[(size_t)gk * HH + gc] : 0.0f;
    }
    __syncthreads();
    do_mma(Asb[0], Bsb[0]);
  }
#else
  for (int k0 = ks; k0 < ke; k0 += 32) {
    __syncthreads();
    for (int idx = tid; idx < 16 * 32; idx += 128) {
      int r = idx >> 5, c = idx & 31, gk = k0 + c;
      Asb[0][idx] = (gk < ke) ? inputs[(size_t)(row0 + r) * NN + gk] : 0.0f;
    }
    for (int idx = tid; idx < 32 * 128; idx += 128) {
      int kk = idx >> 7, c = idx & 127, gk = k0 + kk, gc = c0 + c;
      Bsb[0][kk * BLDN + c] = (gk < ke && gc < HH) ? W1[(size_t)gk * HH + gc] : 0.0f;
    }
    __syncthreads();
    do_mma(Asb[0], Bsb[0]);
  }
#endif

  float* outp = hpart + (size_t)s * BB * HP_STRIDE;
  const int rbase = (lane >> 4) * 8;
#pragma unroll
  for (int t = 0; t < 2; ++t) {
    f32x8 a = t ? acc1 : acc0;
    int col = c0 + wave * 32 + t * 16 + (lane & 15);
    if (col < HH) {
#pragma unroll
      for (int v = 0; v < 8; ++v)
        outp[(size_t)(row0 + rbase + v) * HP_STRIDE + col] = a[v];
    }
  }
}

// ------- 4) encoder epilogue: reduce splits + p@W1b + tanh(.*xinv + b1) ------

__global__ __launch_bounds__(640) void k_enc_epi(const float* __restrict__ hpart,
                                                 const float* __restrict__ p,
                                                 const float* __restrict__ xinv,
                                                 const float* __restrict__ W1,
                                                 const float* __restrict__ b1,
                                                 float* __restrict__ h) {
  const int b = blockIdx.x;
  const int j = threadIdx.x;
  if (j >= HH) return;
  float acc = 0.0f;
  for (int s = 0; s < SPLITK_E; ++s)
    acc += hpart[((size_t)s * BB + b) * HP_STRIDE + j];
  const float* pb = p + (size_t)b * DD;
#pragma unroll 8
  for (int d = 0; d < DD; ++d)
    acc += pb[d] * W1[(size_t)(NN + d) * HH + j];
  h[(size_t)b * HH + j] = tanhf(acc * xinv[b] + b1[j]);
}

// ------- 5) z_mean / z_log_var / z -------------------------------------------

__global__ __launch_bounds__(256) void k_z(const float* __restrict__ h,
                                           const float* __restrict__ Wm,
                                           const float* __restrict__ bm,
                                           const float* __restrict__ Wv,
                                           const float* __restrict__ bv,
                                           const float* __restrict__ eps,
                                           float* __restrict__ zm,
                                           float* __restrict__ zlv,
                                           float* __restrict__ z) {
  const int b = blockIdx.x;
  const int l = threadIdx.x;
  if (l >= LL) return;
  float am = 0.0f, av = 0.0f;
  for (int k = 0; k < HH; ++k) {
    float hk = h[(size_t)b * HH + k];
    am += hk * Wm[(size_t)k * LL + l];
    av += hk * Wv[(size_t)k * LL + l];
  }
  am += bm[l];
  av += bv[l];
  zm[(size_t)b * LL + l]  = am;
  zlv[(size_t)b * LL + l] = av;
  z[(size_t)b * LL + l]   = am + expf(0.5f * av) * eps[(size_t)b * LL + l];
}

// ------- 6) a1 = tanh(z @ Wd1 + bd1) -----------------------------------------

__global__ __launch_bounds__(640) void k_a1(const float* __restrict__ z,
                                            const float* __restrict__ Wd1,
                                            const float* __restrict__ bd1,
                                            float* __restrict__ a1) {
  const int b = blockIdx.x;
  const int j = threadIdx.x;
  if (j >= HH) return;
  float acc = 0.0f;
#pragma unroll 4
  for (int k = 0; k < LL; ++k)
    acc += z[(size_t)b * LL + k] * Wd1[(size_t)k * HH + j];
  a1[(size_t)b * HH + j] = tanhf(acc + bd1[j]);
}

// ------- 7) decoder GEMM: out = a1 @ Wd2 + bd2 -------------------------------

__global__ __launch_bounds__(128) void k_dec_gemm(const float* __restrict__ a1,
                                                  const float* __restrict__ Wd2,
                                                  const float* __restrict__ bd2,
                                                  float* __restrict__ out) {
  __shared__ alignas(16) float Asb[2][16 * 32];
  __shared__ alignas(16) float Bsb[2][32 * BLDN];
  const int rowT = blockIdx.x;  // fastest -> concurrent blocks share Wd2 tile in L2
  const int cb   = blockIdx.y;
  const int row0 = rowT * 16, c0 = cb * 128;
  const int tid = threadIdx.x, lane = tid & 31, wave = tid >> 5;

  f32x8 acc0 = {}, acc1 = {};
  auto do_mma = [&](const float* Af, const float* Bf) {
    bf16x16 a  = frag_a_f32(Af, lane);
    bf16x16 b0 = frag_b_f32(Bf, BLDN, lane, wave * 32);
    bf16x16 b1 = frag_b_f32(Bf, BLDN, lane, wave * 32 + 16);
    acc0 = __builtin_amdgcn_wmma_f32_16x16x32_bf16(false, a, false, b0, (short)0,
                                                   acc0, false, false);
    acc1 = __builtin_amdgcn_wmma_f32_16x16x32_bf16(false, a, false, b1, (short)0,
                                                   acc1, false, false);
  };

  // Last column block may over-read past the end of Wd2 with async b128 ->
  // use the guarded sync path there (32 of 12512 blocks).
  bool useAsync = (cb + 1 < (int)gridDim.y);
#if HAVE_ASYNC
  if (useAsync) {
    const int nfull = HH >> 5;  // 18 full steps, tail 24
    auto issue = [&](int buf, int k0) {
      int r = tid >> 3, c4 = (tid & 7) * 4;
      async_ld_b128(a1 + (size_t)(row0 + r) * HH + k0 + c4, &Asb[buf][r * 32 + c4]);
#pragma unroll
      for (int q = 0; q < 8; ++q) {
        int idx = q * 128 + tid;
        int kk = idx >> 5, cc = (idx & 31) * 4;
        async_ld_b128(Wd2 + (size_t)(k0 + kk) * NN + c0 + cc, &Bsb[buf][kk * BLDN + cc]);
      }
    };
    issue(0, 0);
    for (int i = 0; i < nfull; ++i) {
      __syncthreads();
      if (i + 1 < nfull) {
        issue((i + 1) & 1, (i + 1) * 32);
        __builtin_amdgcn_s_wait_asynccnt(9);
      } else {
        __builtin_amdgcn_s_wait_asynccnt(0);
      }
      __syncthreads();
      do_mma(Asb[i & 1], Bsb[i & 1]);
    }
    {  // tail: k = 576..599
      const int k0 = nfull * 32;
      __syncthreads();
      for (int idx = tid; idx < 16 * 32; idx += 128) {
        int r = idx >> 5, c = idx & 31, gk = k0 + c;
        Asb[0][idx] = (gk < HH) ? a1[(size_t)(row0 + r) * HH + gk] : 0.0f;
      }
      for (int idx = tid; idx < 32 * 128; idx += 128) {
        int kk = idx >> 7, c = idx & 127, gk = k0 + kk;
        Bsb[0][kk * BLDN + c] = (gk < HH) ? Wd2[(size_t)gk * NN + c0 + c] : 0.0f;
      }
      __syncthreads();
      do_mma(Asb[0], Bsb[0]);
    }
  } else
#endif
  {
    for (int k0 = 0; k0 < HH; k0 += 32) {
      __syncthreads();
      for (int idx = tid; idx < 16 * 32; idx += 128) {
        int r = idx >> 5, c = idx & 31, gk = k0 + c;
        Asb[0][idx] = (gk < HH) ? a1[(size_t)(row0 + r) * HH + gk] : 0.0f;
      }
      for (int idx = tid; idx < 32 * 128; idx += 128) {
        int kk = idx >> 7, c = idx & 127, gk = k0 + kk, gc = c0 + c;
        Bsb[0][kk * BLDN + c] = (gk < HH && gc < NN) ? Wd2[(size_t)gk * NN + gc] : 0.0f;
      }
      __syncthreads();
      do_mma(Asb[0], Bsb[0]);
    }
  }

  const int rbase = (lane >> 4) * 8;
#pragma unroll
  for (int t = 0; t < 2; ++t) {
    f32x8 a = t ? acc1 : acc0;
    int col = c0 + wave * 32 + t * 16 + (lane & 15);
    if (col < NN) {
      float bias = bd2[col];
#pragma unroll
      for (int v = 0; v < 8; ++v)
        out[(size_t)(row0 + rbase + v) * NN + col] = a[v] + bias;
    }
  }
}

// ------- 8) KL loss (deterministic two-stage reduction) ----------------------

__global__ __launch_bounds__(1024) void k_kl1(const float* __restrict__ zm,
                                              const float* __restrict__ zlv,
                                              float* __restrict__ part) {
  __shared__ float red[1024];
  const int idx = blockIdx.x * 1024 + threadIdx.x;  // 102400 exactly
  float m = zm[idx], v = zlv[idx];
  red[threadIdx.x] = v - m * m - expf(v) + 1.0f;
  __syncthreads();
  for (int off = 512; off > 0; off >>= 1) {
    if (threadIdx.x < off) red[threadIdx.x] += red[threadIdx.x + off];
    __syncthreads();
  }
  if (threadIdx.x == 0) part[blockIdx.x] = red[0];
}

__global__ void k_kl2(const float* __restrict__ part, float* __restrict__ out_kl) {
  if (threadIdx.x == 0) {
    float t = 0.0f;
    for (int i = 0; i < 100; ++i) t += part[i];  // fixed order -> deterministic
    *out_kl = -0.5f * t / (float)(BB * LL);
  }
}

// ---------------- launch -----------------------------------------------------

extern "C" void kernel_launch(void* const* d_in, const int* in_sizes, int n_in,
                              void* d_out, int out_size, void* d_ws, size_t ws_size,
                              hipStream_t stream) {
  const float* inputs   = (const float*)d_in[0];
  const float* item_emb = (const float*)d_in[1];
  const float* W1  = (const float*)d_in[2];
  const float* b1  = (const float*)d_in[3];
  const float* Wm  = (const float*)d_in[4];
  const float* bm  = (const float*)d_in[5];
  const float* Wv  = (const float*)d_in[6];
  const float* bv  = (const float*)d_in[7];
  const float* Wd1 = (const float*)d_in[8];
  const float* bd1 = (const float*)d_in[9];
  const float* Wd2 = (const float*)d_in[10];
  const float* bd2 = (const float*)d_in[11];
  const float* eps = (const float*)d_in[12];
  float* out = (float*)d_out;

  float* ws = (float*)d_ws;
  size_t off = 0;
  float* wsPool = ws + off; off += (size_t)SPLITK_P * BB * POOL_STRIDE;
  float* wsP    = ws + off; off += (size_t)BB * DD;
  float* wsXinv = ws + off; off += BB;
  float* wsHp   = ws + off; off += (size_t)SPLITK_E * BB * HP_STRIDE;
  float* wsH    = ws + off; off += (size_t)BB * HH;
  float* wsZm   = ws + off; off += (size_t)BB * LL;
  float* wsZlv  = ws + off; off += (size_t)BB * LL;
  float* wsZ    = ws + off; off += (size_t)BB * LL;
  float* wsA1   = ws + off; off += (size_t)BB * HH;
  float* wsKl   = ws + off; off += 128;

  dim3 gp(BB / 16, SPLITK_P);
  k_pool<<<gp, 128, 0, stream>>>(inputs, item_emb, wsPool);
  k_pool_fin<<<BB, 64, 0, stream>>>(wsPool, wsP, wsXinv);
  dim3 ge(BB / 16, (HH + 127) / 128, SPLITK_E);
  k_enc_gemm<<<ge, 128, 0, stream>>>(inputs, W1, wsHp);
  k_enc_epi<<<BB, 640, 0, stream>>>(wsHp, wsP, wsXinv, W1, b1, wsH);
  k_z<<<BB, 256, 0, stream>>>(wsH, Wm, bm, Wv, bv, eps, wsZm, wsZlv, wsZ);
  k_a1<<<BB, 640, 0, stream>>>(wsZ, Wd1, bd1, wsA1);
  dim3 gd(BB / 16, (NN + 127) / 128);
  k_dec_gemm<<<gd, 128, 0, stream>>>(wsA1, Wd2, bd2, out);
  k_kl1<<<(BB * LL) / 1024, 1024, 0, stream>>>(wsZm, wsZlv, wsKl);
  k_kl2<<<1, 32, 0, stream>>>(wsKl, out + (size_t)BB * NN);
}
